// AUwGCNWithGATAndTCN_42511586296092
// MI455X (gfx1250) — matrix operand order repaired and checked by reading, not compiled, not run on hardware
//
#include <hip/hip_runtime.h>

typedef __attribute__((ext_vector_type(16))) _Float16 v16h;
typedef __attribute__((ext_vector_type(8)))  float    v8f;

#define T_LEN 2048
#define NBATCH 16
#define NNODE 12
#define BINST (NBATCH * T_LEN)   // 32768 graph instances

// A-fragment K-index pattern for 16-bit 16x32 A (per CDNA5 ISA 7.12.2):
// two contiguous 8-element runs: [8h, 8h+8) and [16+8h, 16+8h+8)
__device__ __forceinline__ int a_koff(int e, int h) {
    int v = e >> 1, p = e & 1;
    return ((v & 3) * 2 + p) + 8 * h + (v >> 2) * 16;
}

// ---------------------------------------------------------------------------
// Stage 1: GraphConvolution 2->16 + linear residual + relu  (per element)
// ---------------------------------------------------------------------------
__global__ void gc_kernel(const float* __restrict__ x, const float* __restrict__ adj,
                          const float* __restrict__ gcw, const float* __restrict__ gcb,
                          const float* __restrict__ gcr, float* __restrict__ hpre) {
    int gid = blockIdx.x * blockDim.x + threadIdx.x;
    if (gid >= BINST * NNODE * 16) return;
    int f = gid & 15;
    int n = (gid >> 4) % NNODE;
    int B = gid / (NNODE * 16);
    const float* xb = x + B * (NNODE * 2);
    float acc = gcb[f];
    #pragma unroll
    for (int j = 0; j < NNODE; ++j) {
        float sup = xb[j * 2] * gcw[f] + xb[j * 2 + 1] * gcw[16 + f];
        acc += adj[n * 12 + j] * sup;
    }
    acc += xb[n * 2] * gcr[f] + xb[n * 2 + 1] * gcr[16 + f];
    hpre[gid] = fmaxf(acc, 0.f);
}

// ---------------------------------------------------------------------------
// Per-channel sum / sumsq reduction for training-mode BN
// element addr = o*ostride + c*cstride + i*istride ;  stats[c]=sum, stats[C+c]=sumsq
// ---------------------------------------------------------------------------
__global__ void channel_reduce(const float* __restrict__ p, float* __restrict__ stats,
                               int C, int O, int I, int ostride, int cstride, int istride) {
    __shared__ float ss[256], sq[256];
    int c = blockIdx.x;
    float s = 0.f, q = 0.f;
    int total = O * I;
    for (int idx = threadIdx.x; idx < total; idx += blockDim.x) {
        int o = idx / I, i = idx % I;
        float v = p[(long)o * ostride + (long)c * cstride + (long)i * istride];
        s += v; q += v * v;
    }
    ss[threadIdx.x] = s; sq[threadIdx.x] = q;
    __syncthreads();
    for (int st = 128; st > 0; st >>= 1) {
        if ((int)threadIdx.x < st) {
            ss[threadIdx.x] += ss[threadIdx.x + st];
            sq[threadIdx.x] += sq[threadIdx.x + st];
        }
        __syncthreads();
    }
    if (threadIdx.x == 0) { stats[c] = ss[0]; stats[C + c] = sq[0]; }
}

// stats layout: [sum(C), sumsq(C), scale(C), shift(C)]
__global__ void bn_finalize(float* __restrict__ stats, const float* __restrict__ g,
                            const float* __restrict__ b, int C, float invcnt) {
    int c = threadIdx.x;
    if (c < C) {
        float mean = stats[c] * invcnt;
        float var  = stats[C + c] * invcnt - mean * mean;
        float sc   = g[c] * rsqrtf(var + 1e-5f);
        stats[2 * C + c] = sc;
        stats[3 * C + c] = b[c] - mean * sc;
    }
}

// ---------------------------------------------------------------------------
// Stage 2: bn1-apply + GAT + TCN conv (per 12x16 instance; block = 192 thr)
// ---------------------------------------------------------------------------
__global__ void gat_tcn_kernel(const float* __restrict__ hpre, const float* __restrict__ bn1st,
                               const float* __restrict__ gatw, const float* __restrict__ tcnw,
                               const float* __restrict__ tcnb,
                               float* __restrict__ h2out, float* __restrict__ yout) {
    __shared__ float hs[NNODE][16];
    __shared__ float hp[NNODE][16];
    __shared__ float E[NNODE][NNODE];
    __shared__ float colm[NNODE], cold[NNODE], srow[NNODE];
    int tid = threadIdx.x;
    int B = blockIdx.x;
    int n = tid / 16, f = tid % 16;
    const float* bsc = bn1st + 32;
    const float* bsh = bn1st + 48;
    float v = hpre[B * 192 + tid];
    hs[n][f] = fmaxf(bsc[f] * v + bsh[f], 0.f);
    __syncthreads();
    float acc = 0.f;
    #pragma unroll
    for (int g = 0; g < 16; ++g) acc += hs[n][g] * gatw[g * 16 + f];
    hp[n][f] = acc;
    __syncthreads();
    if (tid < 144) {
        int i = tid / 12, j = tid % 12;
        float d = 0.f;
        #pragma unroll
        for (int k = 0; k < 16; ++k) d += hp[i][k] * hp[j][k];
        E[i][j] = d > 0.f ? d : 0.2f * d;
    }
    __syncthreads();
    if (tid < 12) {      // softmax over axis=1 (rows i) per column j
        float m = -3e38f;
        for (int i = 0; i < 12; ++i) m = fmaxf(m, E[i][tid]);
        float den = 0.f;
        for (int i = 0; i < 12; ++i) den += __expf(E[i][tid] - m);
        colm[tid] = m; cold[tid] = den;
    }
    __syncthreads();
    if (tid < 12) {      // s[i] = sum_j att[i][j]
        float s = 0.f;
        for (int j = 0; j < 12; ++j) s += __expf(E[tid][j] - colm[j]) / cold[j];
        srow[tid] = s;
    }
    __syncthreads();
    float h2 = hp[n][f] * srow[n];
    h2out[B * 192 + n * 16 + f] = h2;
    __syncthreads();
    hs[n][f] = h2;
    __syncthreads();
    // TCN conv over node axis, k=3 pad=1 : y[f][n]
    float y = tcnb[f];
    #pragma unroll
    for (int k = 0; k < 3; ++k) {
        int nn = n + k - 1;
        if (nn >= 0 && nn < 12) {
            #pragma unroll
            for (int g = 0; g < 16; ++g) y += tcnw[f * 48 + g * 3 + k] * hs[nn][g];
        }
    }
    yout[B * 192 + f * 12 + n] = y;
}

// ---------------------------------------------------------------------------
// Stage 3: tcn-bn apply + relu + residual relu, reshape to z0 (b,192,t)
// ---------------------------------------------------------------------------
__global__ void z0_kernel(const float* __restrict__ y, const float* __restrict__ h2,
                          const float* __restrict__ tcnst, float* __restrict__ z0) {
    int gid = blockIdx.x * blockDim.x + threadIdx.x;
    if (gid >= BINST * 192) return;
    int f = gid & 15;
    int n = (gid >> 4) % 12;
    int B = gid / 192;
    const float* sc = tcnst + 32;
    const float* sh = tcnst + 48;
    float yv = y[B * 192 + f * 12 + n];
    yv = fmaxf(sc[f] * yv + sh[f], 0.f);
    float hc = fmaxf(yv + h2[B * 192 + n * 16 + f], 0.f);
    int b = B / T_LEN, t = B % T_LEN;
    z0[((long)b * 192 + n * 16 + f) * T_LEN + t] = hc;
}

// ---------------------------------------------------------------------------
// Generic 1D conv as implicit GEMM via WMMA f16 (fp32 accumulate), templated
// on shape so all index math is compile-time (no integer-division sequences,
// accumulators guaranteed in VGPRs, loops fully unrolled).
// Block = 128 thr (4 waves) covers 64 t-positions and all Cout tiles.
// Input is staged as an im2col tile im[t][q] (q = k*Cin+ci) in f16 with
// BN(scale/shift)+relu fused and zero padding; with this layout the WMMA A
// fragment is two contiguous 16B-aligned 8-half runs -> 2x ds_load_b128, and
// the weight B fragment is two contiguous 8-half runs (same k order as A).
// ---------------------------------------------------------------------------
template <int CIN, int COUT, int K, int DIL>
__global__ void conv_wmma_kernel(
    const float* __restrict__ in, const float* __restrict__ w,
    const float* __restrict__ in_scale, const float* __restrict__ in_shift, int in_relu,
    const float* __restrict__ res, const float* __restrict__ res_scale,
    const float* __restrict__ res_shift, int res_relu,
    float* __restrict__ out, int pad)
{
    constexpr int T      = T_LEN;
    constexpr int CINK   = CIN * K;
    constexpr int KQ     = (CINK + 31) & ~31;
    constexpr int KQP    = KQ + 8;                 // padded LDS row stride (16B mult)
    constexpr int COUT16 = (COUT + 15) & ~15;
    constexpr int NT     = COUT16 / 16;
    constexpr int NK     = KQ / 32;
    extern __shared__ _Float16 smem[];
    _Float16* Wl = smem;                           // [COUT16][KQP]
    _Float16* im = smem + COUT16 * KQP;            // [64][KQP] im2col tile
    int b  = blockIdx.y;
    int t0 = blockIdx.x * 64;
    int tid = threadIdx.x;

    // stage f16 weights: Wl[co][q], q = k*CIN + ci
    for (int idx = tid; idx < COUT16 * KQ; idx += 128) {
        int co = idx / KQ, q = idx % KQ;
        float val = 0.f;
        if (co < COUT && q < CINK) {
            int k = q / CIN, ci = q % CIN;
            val = w[(co * CIN + ci) * K + k];
        }
        Wl[co * KQP + q] = (_Float16)val;
    }
    // im2col staging (t fastest -> coalesced global reads); pad -> 0
    for (int idx = tid; idx < 64 * KQ; idx += 128) {
        int q = idx >> 6, tt = idx & 63;
        float v = 0.f;
        if (q < CINK) {
            int k = q / CIN, ci = q % CIN;
            int t = t0 + tt + k * DIL - pad;
            if (t >= 0 && t < T) {
                v = in[(b * CIN + ci) * T + t];
                if (in_scale) v = in_scale[ci] * v + in_shift[ci];
                if (in_relu)  v = fmaxf(v, 0.f);
            }
        }
        im[tt * KQP + q] = (_Float16)v;
    }
    __syncthreads();

    int lane = tid & 31, wid = tid >> 5;
    int l15 = lane & 15, h = lane >> 4;
    const _Float16* arow = im + (wid * 16 + l15) * KQP;
    v8f acc[NT];
    #pragma unroll
    for (int i = 0; i < NT; ++i) acc[i] = (v8f){0.f,0.f,0.f,0.f,0.f,0.f,0.f,0.f};

    #pragma unroll
    for (int kk = 0; kk < NK; ++kk) {
        const int q0 = kk * 32;
        v16h a;   // two contiguous 8-half runs (interleaved A K-pattern)
        #pragma unroll
        for (int e = 0; e < 8; ++e) a[e] = arow[q0 + 8 * h + e];
        #pragma unroll
        for (int e = 0; e < 8; ++e) a[8 + e] = arow[q0 + 16 + 8 * h + e];
        #pragma unroll
        for (int nt = 0; nt < NT; ++nt) {
            const _Float16* brow = Wl + (nt * 16 + l15) * KQP + q0 + 8 * h;
            v16h bf;  // two contiguous 8-half runs (same K-pattern as A)
            #pragma unroll
            for (int e = 0; e < 16; ++e) bf[e] = brow[e + (e >= 8 ? 8 : 0)];
            acc[nt] = __builtin_amdgcn_wmma_f32_16x16x32_f16(
                false, a, false, bf, (short)0, acc[nt], false, false);
        }
    }
    // epilogue: D lane layout = (col co = l15, row t = r + 8h)
    #pragma unroll
    for (int nt = 0; nt < NT; ++nt) {
        int co = nt * 16 + l15;
        if (co >= COUT) continue;
        #pragma unroll
        for (int r = 0; r < 8; ++r) {
            int t = t0 + wid * 16 + r + 8 * h;
            float val = acc[nt][r];
            if (res) {
                float rv = res[(b * COUT + co) * T + t];
                if (res_scale) rv = res_scale[co] * rv + res_shift[co];
                if (res_relu)  rv = fmaxf(rv, 0.f);
                val += rv;
            }
            out[(b * COUT + co) * T + t] = val;
        }
    }
}

// ---------------------------------------------------------------------------
// NonLocal block, flash-attention style. d=32, softmax over j, never
// materializes the 268MB (b,t,t) score matrix.
// Block = 128 thr; each wave owns 16 queries; K/V staged in LDS per 32-j tile
// (with global_prefetch of the next tile). K tile is stored TRANSPOSED
// ([j][c], 80B row stride) and V tile row-padded to 80B so every WMMA A
// fragment is two contiguous 16B-aligned 8-half runs (ds_load_b128, no
// scattered u16 loads). Scores computed transposed (S_T[j,i]) so softmax
// stats are in-lane + one shfl_xor(16); exp(P) serves as the PV B-fragment
// as-is, without cross-lane exchange.
// ---------------------------------------------------------------------------
__global__ void attn_kernel(const float* __restrict__ Kmat,   // th  (keys, j)
                            const float* __restrict__ Qmat,   // ph  (queries, i)
                            const float* __restrict__ Vmat,   // gg
                            float* __restrict__ Omat) {
    const int T = T_LEN, D = 32;
    __shared__ alignas(16) _Float16 KtT[32][40];   // [j][c], 80B stride
    __shared__ alignas(16) _Float16 Vt[32][40];    // [c][j], 80B stride
    int b = blockIdx.y;
    int tid = threadIdx.x, lane = tid & 31, wid = tid >> 5;
    int h = lane >> 4, l15 = lane & 15;
    int ig = blockIdx.x * 64 + wid * 16 + l15;       // this lane's query column i
    const long base = (long)b * D * T;

    v16h bq;                                          // Q fragment (loop-invariant)
    #pragma unroll
    for (int e = 0; e < 16; ++e)
        bq[e] = (_Float16)Qmat[base + (long)(e + 8 * h + (e >= 8 ? 8 : 0)) * T + ig];

    const v8f vzero = (v8f){0.f,0.f,0.f,0.f,0.f,0.f,0.f,0.f};
    v8f o1 = vzero, o2 = vzero;
    float mstate = -3e38f, lstate = 0.f;

    for (int jb = 0; jb < T; jb += 32) {
        __syncthreads();
        for (int idx = tid; idx < 32 * 32; idx += 128) {
            int c = idx >> 5, j = idx & 31;          // j fastest -> coalesced global
            long ka = base + (long)c * T + jb + j;
            KtT[j][c] = (_Float16)Kmat[ka];          // transposed store
            Vt[c][j]  = (_Float16)Vmat[ka];
            __builtin_prefetch(&Kmat[ka + 32], 0, 0);  // next 32-j tile
            __builtin_prefetch(&Vmat[ka + 32], 0, 0);
        }
        __syncthreads();
        // S_T tiles: rows j (two 16-row tiles), cols i ; K-dim = c (32)
        // A fragment = two contiguous 8-half runs of KtT row j
        v16h a1, a2;
        #pragma unroll
        for (int e = 0; e < 8; ++e) {
            a1[e]     = KtT[l15][8 * h + e];
            a1[e + 8] = KtT[l15][16 + 8 * h + e];
            a2[e]     = KtT[16 + l15][8 * h + e];
            a2[e + 8] = KtT[16 + l15][16 + 8 * h + e];
        }
        v8f s1 = __builtin_amdgcn_wmma_f32_16x16x32_f16(false, a1, false, bq, (short)0, vzero, false, false);
        v8f s2 = __builtin_amdgcn_wmma_f32_16x16x32_f16(false, a2, false, bq, (short)0, vzero, false, false);
        // online softmax over j (per column i)
        float tm = -3e38f;
        #pragma unroll
        for (int r = 0; r < 8; ++r) tm = fmaxf(tm, fmaxf(s1[r], s2[r]));
        tm = fmaxf(tm, __shfl_xor(tm, 16, 32));
        float mnew = fmaxf(mstate, tm);
        float alpha = __expf(mstate - mnew);
        float p1[8], p2[8], psum = 0.f;
        #pragma unroll
        for (int r = 0; r < 8; ++r) {
            p1[r] = __expf(s1[r] - mnew);
            p2[r] = __expf(s2[r] - mnew);
            psum += p1[r] + p2[r];
        }
        psum += __shfl_xor(psum, 16, 32);
        lstate = lstate * alpha + psum;
        mstate = mnew;
        #pragma unroll
        for (int r = 0; r < 8; ++r) { o1[r] *= alpha; o2[r] *= alpha; }
        // P (C-layout) is already the P*V B-fragment (k = j order matches V); shfl result unused
        v16h bp;
        #pragma unroll
        for (int r = 0; r < 8; ++r) {
            float give = h ? p1[r] : p2[r];
            float got  = __shfl_xor(give, 16, 32);
            bp[r]     = (_Float16)p1[r];
            bp[r + 8] = (_Float16)p2[r];
        }
        // V fragments: rows c, K-dim = j (32); contiguous aligned runs
        v16h av1, av2;
        #pragma unroll
        for (int e = 0; e < 8; ++e) {
            av1[e]     = Vt[l15][8 * h + e];
            av1[e + 8] = Vt[l15][16 + 8 * h + e];
            av2[e]     = Vt[16 + l15][8 * h + e];
            av2[e + 8] = Vt[16 + l15][16 + 8 * h + e];
        }
        o1 = __builtin_amdgcn_wmma_f32_16x16x32_f16(false, av1, false, bp, (short)0, o1, false, false);
        o2 = __builtin_amdgcn_wmma_f32_16x16x32_f16(false, av2, false, bp, (short)0, o2, false, false);
    }
    float inv = 1.f / lstate;
    #pragma unroll
    for (int r = 0; r < 8; ++r) {
        Omat[base + (long)(r + 8 * h) * T + ig]      = o1[r] * inv;
        Omat[base + (long)(16 + r + 8 * h) * T + ig] = o2[r] * inv;
    }
}

// ---------------------------------------------------------------------------
extern "C" void kernel_launch(void* const* d_in, const int* in_sizes, int n_in,
                              void* d_out, int out_size, void* d_ws, size_t ws_size,
                              hipStream_t stream) {
    const float* x        = (const float*)d_in[0];
    const float* adj      = (const float*)d_in[1];
    const float* gc_w     = (const float*)d_in[2];
    const float* gc_b     = (const float*)d_in[3];
    const float* gc_res_w = (const float*)d_in[4];
    const float* bn1_g    = (const float*)d_in[5];
    const float* bn1_b    = (const float*)d_in[6];
    const float* gat_w    = (const float*)d_in[7];
    const float* tcn_w    = (const float*)d_in[8];
    const float* tcn_b    = (const float*)d_in[9];
    const float* tcn_bn_g = (const float*)d_in[10];
    const float* tcn_bn_b = (const float*)d_in[11];
    const float* s1_w     = (const float*)d_in[12];
    const float* s1_g     = (const float*)d_in[13];
    const float* s1_b     = (const float*)d_in[14];
    const float* s2_w     = (const float*)d_in[15];
    const float* s2_g     = (const float*)d_in[16];
    const float* s2_b     = (const float*)d_in[17];
    const float* s3_w     = (const float*)d_in[18];
    const float* s3_g     = (const float*)d_in[19];
    const float* s3_b     = (const float*)d_in[20];
    const float* nl_theta = (const float*)d_in[21];
    const float* nl_phi   = (const float*)d_in[22];
    const float* nl_g     = (const float*)d_in[23];
    const float* nl_out_w = (const float*)d_in[24];
    const float* cls_w    = (const float*)d_in[25];
    float* ws = (float*)d_ws;

    // workspace layout (floats), with buffer reuse
    const long OFF_HPRE = 0L;            // 6291456  (dead after GAT)
    const long OFF_H2   = 6291456L;      // 6291456  (dead after z0)
    const long OFF_Y    = 12582912L;     // 6291456  (dead after z0)
    const long OFF_Z0   = 0L;            // reuse HPRE
    const long OFF_Z1   = 18874368L;     // 2097152
    const long OFF_Z2   = 20971520L;
    const long OFF_Z3   = 23068672L;
    const long OFF_TH   = 25165824L;     // 1048576
    const long OFF_PH   = 26214400L;
    const long OFF_GG   = 27262976L;
    const long OFF_O    = OFF_H2;        // reuse
    const long OFF_ZF   = OFF_Y;         // reuse
    const long OFF_ST   = 28311552L;
    float* ST_BN1 = ws + OFF_ST;         // C=16 : [sum16,sq16,scale16,shift16]
    float* ST_TCN = ws + OFF_ST + 256;   // C=16
    float* ST_S1  = ws + OFF_ST + 512;   // C=64 : scale @+128, shift @+192
    float* ST_S2  = ws + OFF_ST + 768;
    float* ST_S3  = ws + OFF_ST + 1024;

    auto smemsz = [](int Cin, int Cout, int K) -> size_t {
        int Kq = ((Cin * K) + 31) & ~31;
        int Kqp = Kq + 8;
        int Cout16 = (Cout + 15) & ~15;
        return (size_t)(Cout16 + 64) * Kqp * 2;
    };
    dim3 gridc(T_LEN / 64, NBATCH);

    // graph stage
    gc_kernel<<<(BINST * 192 + 255) / 256, 256, 0, stream>>>(x, adj, gc_w, gc_b, gc_res_w, ws + OFF_HPRE);
    channel_reduce<<<16, 256, 0, stream>>>(ws + OFF_HPRE, ST_BN1, 16, BINST, 12, 192, 1, 16);
    bn_finalize<<<1, 64, 0, stream>>>(ST_BN1, bn1_g, bn1_b, 16, 1.f / (32768.f * 12.f));
    gat_tcn_kernel<<<BINST, 192, 0, stream>>>(ws + OFF_HPRE, ST_BN1, gat_w, tcn_w, tcn_b,
                                              ws + OFF_H2, ws + OFF_Y);
    channel_reduce<<<16, 256, 0, stream>>>(ws + OFF_Y, ST_TCN, 16, BINST, 12, 192, 12, 1);
    bn_finalize<<<1, 64, 0, stream>>>(ST_TCN, tcn_bn_g, tcn_bn_b, 16, 1.f / (32768.f * 12.f));
    z0_kernel<<<(BINST * 192 + 255) / 256, 256, 0, stream>>>(ws + OFF_Y, ws + OFF_H2, ST_TCN, ws + OFF_Z0);

    // sequential conv stack (WMMA implicit GEMM)
    conv_wmma_kernel<192, 64, 1, 1><<<gridc, 128, smemsz(192, 64, 1), stream>>>(
        ws + OFF_Z0, s1_w, nullptr, nullptr, 0, nullptr, nullptr, nullptr, 0,
        ws + OFF_Z1, 0);
    channel_reduce<<<64, 256, 0, stream>>>(ws + OFF_Z1, ST_S1, 64, 16, 2048, 64 * 2048, 2048, 1);
    bn_finalize<<<1, 64, 0, stream>>>(ST_S1, s1_g, s1_b, 64, 1.f / 32768.f);

    conv_wmma_kernel<64, 64, 3, 1><<<gridc, 128, smemsz(64, 64, 3), stream>>>(
        ws + OFF_Z1, s2_w, ST_S1 + 128, ST_S1 + 192, 1, nullptr, nullptr, nullptr, 0,
        ws + OFF_Z2, 1);
    channel_reduce<<<64, 256, 0, stream>>>(ws + OFF_Z2, ST_S2, 64, 16, 2048, 64 * 2048, 2048, 1);
    bn_finalize<<<1, 64, 0, stream>>>(ST_S2, s2_g, s2_b, 64, 1.f / 32768.f);

    conv_wmma_kernel<64, 64, 3, 2><<<gridc, 128, smemsz(64, 64, 3), stream>>>(
        ws + OFF_Z2, s3_w, ST_S2 + 128, ST_S2 + 192, 1, nullptr, nullptr, nullptr, 0,
        ws + OFF_Z3, 2);
    channel_reduce<<<64, 256, 0, stream>>>(ws + OFF_Z3, ST_S3, 64, 16, 2048, 64 * 2048, 2048, 1);
    bn_finalize<<<1, 64, 0, stream>>>(ST_S3, s3_g, s3_b, 64, 1.f / 32768.f);

    // NonLocal: theta/phi/g projections (s3 bn+relu fused on load), flash attention, out-proj+residual
    conv_wmma_kernel<64, 32, 1, 1><<<gridc, 128, smemsz(64, 32, 1), stream>>>(
        ws + OFF_Z3, nl_theta, ST_S3 + 128, ST_S3 + 192, 1, nullptr, nullptr, nullptr, 0,
        ws + OFF_TH, 0);
    conv_wmma_kernel<64, 32, 1, 1><<<gridc, 128, smemsz(64, 32, 1), stream>>>(
        ws + OFF_Z3, nl_phi, ST_S3 + 128, ST_S3 + 192, 1, nullptr, nullptr, nullptr, 0,
        ws + OFF_PH, 0);
    conv_wmma_kernel<64, 32, 1, 1><<<gridc, 128, smemsz(64, 32, 1), stream>>>(
        ws + OFF_Z3, nl_g, ST_S3 + 128, ST_S3 + 192, 1, nullptr, nullptr, nullptr, 0,
        ws + OFF_GG, 0);

    attn_kernel<<<gridc, 128, 0, stream>>>(ws + OFF_TH, ws + OFF_PH, ws + OFF_GG, ws + OFF_O);

    conv_wmma_kernel<32, 64, 1, 1><<<gridc, 128, smemsz(32, 64, 1), stream>>>(
        ws + OFF_O, nl_out_w, nullptr, nullptr, 0,
        ws + OFF_Z3, ST_S3 + 128, ST_S3 + 192, 1,        // residual = relu(bn_s3(z3))
        ws + OFF_ZF, 0);

    // classification head -> d_out (16,10,2048)
    conv_wmma_kernel<64, 10, 3, 2><<<gridc, 128, smemsz(64, 10, 3), stream>>>(
        ws + OFF_ZF, cls_w, nullptr, nullptr, 0, nullptr, nullptr, nullptr, 0,
        (float*)d_out, 2);
}
